// CausalSelfAttention_54503134986501
// MI455X (gfx1250) — matrix operand order repaired and checked by reading, not compiled, not run on hardware
//
#include <hip/hip_runtime.h>
#include <hip/hip_bf16.h>

// ---------------- problem constants ----------------
constexpr int Bb = 4, Tt = 2048, Cc = 1024, Hh = 16, HD = 64;
constexpr int Mrows = Bb * Tt;          // 8192
constexpr int C3 = 3 * Cc;              // 3072

typedef __bf16 bf16;
typedef __attribute__((ext_vector_type(16))) __bf16 v16bf;
typedef __attribute__((ext_vector_type(8)))  __bf16 v8bf;
typedef __attribute__((ext_vector_type(4)))  __bf16 v4bf;
typedef __attribute__((ext_vector_type(8)))  float  v8f;
typedef __attribute__((ext_vector_type(4)))  unsigned int u32x4;
typedef __attribute__((ext_vector_type(8)))  int          i32x8;
typedef __attribute__((ext_vector_type(4)))  int          i32x4;

#define WMMA_BF16(a, b, c) \
  __builtin_amdgcn_wmma_f32_16x16x32_bf16(false, (a), false, (b), (short)0, (c), false, false)

__device__ __forceinline__ v8f v8f_zero() {
  v8f z = {0.f, 0.f, 0.f, 0.f, 0.f, 0.f, 0.f, 0.f};
  return z;
}

// Load a 16x32 bf16 A-fragment row: per-lane elements are k = {0..15}
// relative to p (caller folds in the lane-half offset of 0/16).
__device__ __forceinline__ v16bf ldA(const bf16* p) {
  v8bf lo = *(const v8bf*)(p);
  v8bf hi = *(const v8bf*)(p + 8);
  return __builtin_shufflevector(lo, hi, 0, 1, 2, 3, 4, 5, 6, 7,
                                 8, 9, 10, 11, 12, 13, 14, 15);
}

// ---------------- Tensor Data Mover: 2-D tile Global -> LDS ----------------
// D# per CDNA5 ISA. data_size = 2 bytes (bf16).
// tile_d0 = contiguous elements per line, tile_d1 = lines, stride = line stride (elems).
__device__ __forceinline__ void tdm_load_2d(unsigned int lds_addr, const void* gaddr,
                                            unsigned int tile_d0, unsigned int tile_d1,
                                            unsigned long long stride_elems) {
  unsigned long long ga = (unsigned long long)(uintptr_t)gaddr;
  u32x4 g0;
  g0[0] = 1u;                                           // count=1 (valid), user mode
  g0[1] = lds_addr;                                     // lds_addr (bytes)
  g0[2] = (unsigned int)(ga & 0xffffffffu);             // global_addr[31:0]
  g0[3] = (unsigned int)((ga >> 32) & 0x01ffffffu) | (2u << 30);  // addr[56:32] | type=2

  i32x8 g1;
  g1[0] = (int)(1u << 16);                              // data_size=1 -> 2 bytes
  g1[1] = (int)((tile_d0 & 0xffffu) << 16);             // tensor_dim0[15:0]
  g1[2] = (int)(((tile_d0 >> 16) & 0xffffu) |           // tensor_dim0[31:16]
                ((tile_d1 & 0xffffu) << 16));           // tensor_dim1[15:0]
  g1[3] = (int)(((tile_d1 >> 16) & 0xffffu) |           // tensor_dim1[31:16]
                ((tile_d0 & 0xffffu) << 16));           // tile_dim0
  g1[4] = (int)(tile_d1 & 0xffffu);                     // tile_dim1 (tile_dim2 = 0)
  g1[5] = (int)(unsigned int)(stride_elems & 0xffffffffu);        // dim0_stride[31:0]
  g1[6] = (int)(unsigned int)((stride_elems >> 32) & 0xffffu);    // dim0_stride[47:32]
  g1[7] = 0;

  i32x4 z4 = {0, 0, 0, 0};
#if defined(__clang_major__) && (__clang_major__ >= 23)
  i32x8 z8 = {0, 0, 0, 0, 0, 0, 0, 0};
  __builtin_amdgcn_tensor_load_to_lds(g0, g1, z4, z4, z8, 0);
#else
  __builtin_amdgcn_tensor_load_to_lds(g0, g1, z4, z4, 0);
#endif
}

// ---------------- small prep kernels ----------------

// adjustment[b][t] = 0.1 * -log(prev_probs + 1e-10)
__global__ void bias_kernel(const float* __restrict__ prev, float* __restrict__ adj, int n) {
  int i = blockIdx.x * blockDim.x + threadIdx.x;
  if (i < n) adj[i] = -0.1f * __logf(prev[i] + 1e-10f);
}

// fp32 -> bf16, 4 elements per thread
__global__ void cvt_bf16_vec(const float* __restrict__ src, bf16* __restrict__ dst, int n4) {
  int i = blockIdx.x * blockDim.x + threadIdx.x;
  if (i < n4) {
    float4 v = ((const float4*)src)[i];
    v4bf o;
    o[0] = (bf16)v.x; o[1] = (bf16)v.y; o[2] = (bf16)v.z; o[3] = (bf16)v.w;
    *(v4bf*)(dst + 4 * (size_t)i) = o;
  }
}

// src: K x N fp32 row-major -> dst: N x K bf16 row-major
__global__ void transpose_cvt(const float* __restrict__ src, bf16* __restrict__ dst,
                              int K, int N) {
  __shared__ float tile[32][33];
  const int kb = blockIdx.y * 32, nb = blockIdx.x * 32;
  const int tx = threadIdx.x & 31, ty = threadIdx.x >> 5;   // 32 x 8
#pragma unroll
  for (int r = ty; r < 32; r += 8)
    tile[r][tx] = src[(size_t)(kb + r) * N + nb + tx];
  __syncthreads();
#pragma unroll
  for (int r = ty; r < 32; r += 8)
    dst[(size_t)(nb + r) * K + kb + tx] = (bf16)tile[tx][r];
}

// Per (b,h): V tile (T x 64) -> Vt (64 x T), bf16
__global__ void v_transpose(const bf16* __restrict__ QKV, bf16* __restrict__ Vt) {
  __shared__ bf16 tile[32][64 + 4];
  const int bh = blockIdx.y;
  const int b = bh >> 4, h = bh & 15;
  const int tBase = blockIdx.x * 32;
  const bf16* vsrc = QKV + (size_t)b * Tt * C3 + 2 * Cc + h * HD;
  {
    int d = threadIdx.x & 63, r0 = threadIdx.x >> 6;        // 64 x 4
#pragma unroll
    for (int r = r0; r < 32; r += 4)
      tile[r][d] = vsrc[(size_t)(tBase + r) * C3 + d];
  }
  __syncthreads();
  {
    int key = threadIdx.x & 31, d0 = threadIdx.x >> 5;      // 32 x 8
#pragma unroll
    for (int d = d0; d < 64; d += 8)
      Vt[((size_t)bh * HD + d) * Tt + tBase + key] = tile[key][d];
  }
}

// ---------------- WMMA GEMM with TDM-staged tiles ----------------
// C[M,N] = A[M,K] * Bt[N,K]^T. Block = 256 threads = 8 waves; block tile 128(M) x 64(N).
// Per 64-wide K chunk: TDM stages A(128x64) and B(64x64) into LDS (double buffered,
// issued by wave 0, synced via s_wait_tensorcnt + barrier); waves feed WMMA from ds loads.
template <bool OUT_BF16>
__global__ void __launch_bounds__(256)
wmma_gemm(const bf16* __restrict__ A, const bf16* __restrict__ Bt,
          void* __restrict__ Cptr, int M, int N, int K) {
  const int lane = threadIdx.x & 31;
  const int wave = threadIdx.x >> 5;
  const int l15  = lane & 15;
  const int half = lane >> 4;
  const int mBase = blockIdx.y * 128;
  const int nBase = blockIdx.x * 64;

  __shared__ __align__(128) bf16 ldsA[2][128 * 64];   // 2 x 16 KB
  __shared__ __align__(128) bf16 ldsB[2][64 * 64];    // 2 x  8 KB

  const bf16* gA = A  + (size_t)mBase * K;
  const bf16* gB = Bt + (size_t)nBase * K;

  v8f acc[4];
#pragma unroll
  for (int nt = 0; nt < 4; ++nt) acc[nt] = v8f_zero();

  const int nChunks = K >> 6;  // K/64
  if (wave == 0) {
    tdm_load_2d((unsigned int)(uintptr_t)&ldsA[0][0], gA, 64, 128, (unsigned int)K);
    tdm_load_2d((unsigned int)(uintptr_t)&ldsB[0][0], gB, 64, 64,  (unsigned int)K);
  }

  int buf = 0;
  for (int c = 0; c < nChunks; ++c) {
    if (wave == 0) __builtin_amdgcn_s_wait_tensorcnt(0);   // current buffer landed
    __syncthreads();
    if (wave == 0 && (c + 1) < nChunks) {                  // prefetch next chunk via TDM
      const size_t ko = (size_t)(c + 1) * 64;
      tdm_load_2d((unsigned int)(uintptr_t)&ldsA[buf ^ 1][0], gA + ko, 64, 128, (unsigned int)K);
      tdm_load_2d((unsigned int)(uintptr_t)&ldsB[buf ^ 1][0], gB + ko, 64, 64,  (unsigned int)K);
    }

    const bf16* la = &ldsA[buf][(wave * 16 + l15) * 64 + half * 16];
    const bf16* lb = &ldsB[buf][l15 * 64 + half * 16];
#pragma unroll
    for (int kk = 0; kk < 64; kk += 32) {
      v16bf af = ldA(la + kk);
      v16bf b0 = *(const v16bf*)(lb + 0 * 16 * 64 + kk);
      v16bf b1 = *(const v16bf*)(lb + 1 * 16 * 64 + kk);
      v16bf b2 = *(const v16bf*)(lb + 2 * 16 * 64 + kk);
      v16bf b3 = *(const v16bf*)(lb + 3 * 16 * 64 + kk);
      acc[0] = WMMA_BF16(af, b0, acc[0]);
      acc[1] = WMMA_BF16(af, b1, acc[1]);
      acc[2] = WMMA_BF16(af, b2, acc[2]);
      acc[3] = WMMA_BF16(af, b3, acc[3]);
    }
    __syncthreads();   // everyone done reading buf before it is overwritten
    buf ^= 1;
  }

  const int mOff = half * 8;
  const int mTile = mBase + wave * 16;
  if constexpr (OUT_BF16) {
    bf16* Cb = (bf16*)Cptr;
#pragma unroll
    for (int nt = 0; nt < 4; ++nt)
#pragma unroll
      for (int r = 0; r < 8; ++r)
        Cb[(size_t)(mTile + mOff + r) * N + nBase + nt * 16 + l15] = (bf16)acc[nt][r];
  } else {
    float* Cf = (float*)Cptr;
#pragma unroll
    for (int nt = 0; nt < 4; ++nt)
#pragma unroll
      for (int r = 0; r < 8; ++r)
        Cf[(size_t)(mTile + mOff + r) * N + nBase + nt * 16 + l15] = acc[nt][r];
  }
}

// ---------------- flash attention ----------------
// grid = (T/128, B*H); block = 256 (8 waves); each wave owns 16 queries of one (b,h).
__global__ void __launch_bounds__(256)
flash_attn(const bf16* __restrict__ QKV, const bf16* __restrict__ Vt,
           const float* __restrict__ adj, bf16* __restrict__ Y) {
  const int lane = threadIdx.x & 31;
  const int wave = threadIdx.x >> 5;
  const int l15  = lane & 15;
  const int half = lane >> 4;
  const int mOff = half * 8;
  const int bh = blockIdx.y;
  const int b = bh >> 4, h = bh & 15;
  const int qBase = blockIdx.x * 128 + wave * 16;

  const bf16* Qb = QKV + (size_t)b * Tt * C3 + h * HD;
  const bf16* Kb = Qb + Cc;
  const float* adjb = adj + b * Tt;
  const bf16* Vth = Vt + (size_t)bh * HD * Tt;

  __shared__ __align__(32) bf16 pStage[8][16 * 32];  // per-wave P bounce buffer
  bf16* pBuf = pStage[wave];

  // Q as A-matrix fragments (16 rows x 64 d) -> two K=32 fragments
  const bf16* qp = Qb + (size_t)(qBase + l15) * C3 + half * 16;
  const v16bf aq0 = ldA(qp);
  const v16bf aq1 = ldA(qp + 32);

  v8f o[4];
#pragma unroll
  for (int nt = 0; nt < 4; ++nt) o[nt] = v8f_zero();
  float rowM[8], rowL[8];
#pragma unroll
  for (int r = 0; r < 8; ++r) { rowM[r] = -1e30f; rowL[r] = 0.f; }

  const int qLast = qBase + 15;
  const float scale = 0.125f;  // 1/sqrt(64)

  for (int kc = 0; kc <= qLast; kc += 32) {  // wave-uniform trip count
    const int key0 = kc + l15;
    const int key1 = kc + 16 + l15;
    const float adj0 = adjb[key0];
    const float adj1 = adjb[key1];

    // S = Q * K^T : two 16x16 key tiles, each accumulated over d in two K=32 steps
    v8f s0 = v8f_zero(), s1 = v8f_zero();
    {
      const bf16* kp0 = Kb + (size_t)key0 * C3 + half * 16;
      v16bf bk0 = *(const v16bf*)(kp0);
      v16bf bk1 = *(const v16bf*)(kp0 + 32);
      s0 = WMMA_BF16(aq0, bk0, s0);
      s0 = WMMA_BF16(aq1, bk1, s0);
      const bf16* kp1 = Kb + (size_t)key1 * C3 + half * 16;
      v16bf bk2 = *(const v16bf*)(kp1);
      v16bf bk3 = *(const v16bf*)(kp1 + 32);
      s1 = WMMA_BF16(aq0, bk2, s1);
      s1 = WMMA_BF16(aq1, bk3, s1);
    }

    // online softmax: scale + bias + causal mask, row reductions inside lane halves
#pragma unroll
    for (int r = 0; r < 8; ++r) {
      const int qr = qBase + mOff + r;
      float v0 = s0[r] * scale + adj0; v0 = (key0 <= qr) ? v0 : -1e30f;
      float v1 = s1[r] * scale + adj1; v1 = (key1 <= qr) ? v1 : -1e30f;

      float mx = fmaxf(v0, v1);
      mx = fmaxf(mx, __shfl_xor(mx, 1));
      mx = fmaxf(mx, __shfl_xor(mx, 2));
      mx = fmaxf(mx, __shfl_xor(mx, 4));
      mx = fmaxf(mx, __shfl_xor(mx, 8));

      const float mNew = fmaxf(rowM[r], mx);
      const float corr = __expf(rowM[r] - mNew);
      const float p0 = __expf(v0 - mNew);
      const float p1 = __expf(v1 - mNew);
      float sum = p0 + p1;
      sum += __shfl_xor(sum, 1);
      sum += __shfl_xor(sum, 2);
      sum += __shfl_xor(sum, 4);
      sum += __shfl_xor(sum, 8);

      rowL[r] = rowL[r] * corr + sum;
      rowM[r] = mNew;
      o[0][r] *= corr; o[1][r] *= corr; o[2][r] *= corr; o[3][r] *= corr;

      // stash P in LDS (C/D layout -> row-major 16x32)
      pBuf[(mOff + r) * 32 + l15]      = (bf16)p0;
      pBuf[(mOff + r) * 32 + 16 + l15] = (bf16)p1;
    }
    asm volatile("s_wait_dscnt 0" ::: "memory");

    // reload P in A-matrix layout, then O += P * V
    const v16bf apf = ldA(pBuf + l15 * 32 + half * 16);
#pragma unroll
    for (int nt = 0; nt < 4; ++nt) {
      v16bf bv = *(const v16bf*)(Vth + (size_t)(nt * 16 + l15) * Tt + kc + half * 16);
      o[nt] = WMMA_BF16(apf, bv, o[nt]);
    }
  }

  // normalize and store Y (bf16 activation for the projection GEMM)
  bf16* yp = Y + (size_t)(b * Tt + qBase) * Cc + h * HD;
#pragma unroll
  for (int nt = 0; nt < 4; ++nt)
#pragma unroll
    for (int r = 0; r < 8; ++r) {
      float val = o[nt][r] / rowL[r];
      yp[(size_t)(mOff + r) * Cc + nt * 16 + l15] = (bf16)val;
    }
}

// ---------------- host side ----------------
extern "C" void kernel_launch(void* const* d_in, const int* in_sizes, int n_in,
                              void* d_out, int out_size, void* d_ws, size_t ws_size,
                              hipStream_t stream) {
  const float* x      = (const float*)d_in[0];  // (B,T,C)
  const float* prev   = (const float*)d_in[1];  // (B,T)
  const float* W_attn = (const float*)d_in[2];  // (C,3C)
  const float* W_proj = (const float*)d_in[3];  // (C,C)
  float* out = (float*)d_out;                   // (B,T,C) fp32

  char* ws = (char*)d_ws;
  size_t off = 0;
  auto carve = [&](size_t bytes) -> char* {
    char* p = ws + off;
    off = (off + bytes + 255) & ~(size_t)255;
    return p;
  };
  float* adjP   = (float*)carve((size_t)Bb * Tt * 4);
  bf16*  Xbf    = (bf16*)carve((size_t)Mrows * Cc * 2);
  bf16*  WtAttn = (bf16*)carve((size_t)C3 * Cc * 2);
  bf16*  WtProj = (bf16*)carve((size_t)Cc * Cc * 2);
  bf16*  QKVbf  = (bf16*)carve((size_t)Mrows * C3 * 2);
  bf16*  Vtb    = (bf16*)carve((size_t)Bb * Hh * HD * Tt * 2);
  bf16*  Ybf    = (bf16*)carve((size_t)Mrows * Cc * 2);
  (void)ws_size; (void)in_sizes; (void)n_in; (void)out_size;

  // prep: bias + bf16 conversions / transposes
  bias_kernel<<<(Bb * Tt + 255) / 256, 256, 0, stream>>>(prev, adjP, Bb * Tt);
  {
    int n4 = Mrows * Cc / 4;
    cvt_bf16_vec<<<(n4 + 255) / 256, 256, 0, stream>>>(x, Xbf, n4);
  }
  transpose_cvt<<<dim3(C3 / 32, Cc / 32), 256, 0, stream>>>(W_attn, WtAttn, Cc, C3);
  transpose_cvt<<<dim3(Cc / 32, Cc / 32), 256, 0, stream>>>(W_proj, WtProj, Cc, Cc);

  // QKV = X * W_attn   (bf16 out)
  wmma_gemm<true><<<dim3(C3 / 64, Mrows / 128), 256, 0, stream>>>(
      Xbf, WtAttn, (void*)QKVbf, Mrows, C3, Cc);

  // V transpose per head
  v_transpose<<<dim3(Tt / 32, Bb * Hh), 256, 0, stream>>>(QKVbf, Vtb);

  // flash attention
  flash_attn<<<dim3(Tt / 128, Bb * Hh), 256, 0, stream>>>(QKVbf, Vtb, adjP, Ybf);

  // out = Y * W_proj   (fp32 out)
  wmma_gemm<false><<<dim3(Cc / 64, Mrows / 128), 256, 0, stream>>>(
      Ybf, WtProj, (void*)out, Mrows, Cc, Cc);
}
